// CaDM_S_86346022519231
// MI455X (gfx1250) — matrix operand order repaired and checked by reading, not compiled, not run on hardware
//
#include <hip/hip_runtime.h>
#include <hip/hip_bf16.h>
#include <stdint.h>

typedef __attribute__((ext_vector_type(16))) _Float16 v16h;
typedef __attribute__((ext_vector_type(8)))  float    v8f;
typedef __attribute__((ext_vector_type(4)))  float    v4f;
typedef __attribute__((ext_vector_type(4)))  unsigned int u32x4;
typedef __attribute__((ext_vector_type(8)))  int      i32x8;
typedef __attribute__((ext_vector_type(4)))  int      i32x4;

// ---------------------------------------------------------------------------
// Tensor Data Mover: 2D f32 tile load global->LDS (CDNA5 ISA ch.8 D# layout).
// group0: [1:0]=count=1 | lds_addr | global_addr[56:0] | type=2 ("image")
// group1: data_size=2 (4B), tensor_dim0/1, tile_dim0/1, tensor_dim0_stride
// groups 2/3 (+spare group): zero (<=2D tensor). Issued once by wave 0;
// completion via s_wait_tensorcnt in the issuing wave, then block barrier.
// (6-arg builtin variant: clang-23 / therock toolchain.)
// ---------------------------------------------------------------------------
__device__ __forceinline__ void tdm_load_2d_f32(uint32_t lds_off, const void* gaddr,
                                                uint32_t tile_d0, uint32_t tile_d1,
                                                uint32_t tensor_d0, uint32_t tensor_d1,
                                                uint32_t stride_d0) {
  unsigned long long ga = (unsigned long long)gaddr;
  u32x4 g0;
  g0[0] = 1u;                                      // count=1, user descriptor
  g0[1] = lds_off;                                 // LDS byte address
  g0[2] = (uint32_t)ga;                            // global_addr[31:0]
  g0[3] = (uint32_t)(ga >> 32) | (2u << 30);       // global_addr[56:32] | type=2
  i32x8 g1;
  g1[0] = (int)(2u << 16);                                       // data_size=4B
  g1[1] = (int)((tensor_d0 & 0xFFFFu) << 16);                    // tensor_dim0[15:0]
  g1[2] = (int)(((tensor_d0 >> 16) & 0xFFFFu) |
                ((tensor_d1 & 0xFFFFu) << 16));                  // dim0 hi | dim1 lo
  g1[3] = (int)(((tensor_d1 >> 16) & 0xFFFFu) |
                ((tile_d0 & 0xFFFFu) << 16));                    // dim1 hi | tile_dim0
  g1[4] = (int)(tile_d1 & 0xFFFFu);                              // tile_dim1, tile_dim2=0
  g1[5] = (int)stride_d0;                                        // tensor_dim0_stride lo
  g1[6] = 0;
  g1[7] = 0;
  i32x4 gz  = {0, 0, 0, 0};
  i32x8 gz8 = {0, 0, 0, 0, 0, 0, 0, 0};
  __builtin_amdgcn_tensor_load_to_lds(g0, g1, gz, gz, gz8, 0);
}

// ---------------------------------------------------------------------------
// WMMA fragment layouts (gfx1250 wave32, CDNA5 ISA 7.12.2)
// A 16x32 f16: lane holds row (lane&15); per lane the 16 halfs are the two
//   contiguous runs [kb, kb+8) and [16+kb, 16+kb+8) with kb = (lane>>4)*8.
// B 32x16 f16: lane holds col (lane&15); halfs take k as A does  (lane-contiguous when B is
//   stored K-major -> we pre-transpose weights to f16 [K x N]).
// C/D: vgpr r -> row r + 8*(lane>>4), col = lane&15.
// ---------------------------------------------------------------------------
__device__ __forceinline__ v16h load_a_frag_fast(const float* A, int lda, int r0, int k0) {
  int lane = threadIdx.x & 31;
  int row  = r0 + (lane & 15);
  const float* src = A + (long long)row * lda + k0 + ((lane >> 4) * 8);
  v4f x0 = *(const v4f*)(src);
  v4f x1 = *(const v4f*)(src + 4);
  v4f x2 = *(const v4f*)(src + 16);
  v4f x3 = *(const v4f*)(src + 20);
  v16h a;
#pragma unroll
  for (int i = 0; i < 4; ++i) {
    a[i]      = (_Float16)x0[i];
    a[4 + i]  = (_Float16)x1[i];
    a[8 + i]  = (_Float16)x2[i];
    a[12 + i] = (_Float16)x3[i];
  }
  return a;
}

__device__ __forceinline__ v16h load_b_frag_t(const _Float16* Wt, int ldn, int n0, int k0) {
  int lane = threadIdx.x & 31;
  const _Float16* src = Wt + (long long)k0 * ldn + n0 + (lane & 15);
  int kb = (lane >> 4) * 8;
  v16h b;
#pragma unroll
  for (int s = 0; s < 16; ++s) {
    int k = (s < 8) ? (kb + s) : (16 + kb + (s - 8));
    b[s] = src[(long long)k * ldn];
  }
  return b;
}

// Guarded A fragment (ragged edges only: importance / K=8 spatial MLP)
__device__ __forceinline__ v16h load_a_frag(const float* A, int lda, int r0, int k0,
                                            int Mr, int Kr) {
  int lane = threadIdx.x & 31;
  int row  = r0 + (lane & 15);
  bool okr = row < Mr;
  const float* src = A + (long long)row * lda;
  int kb = (lane >> 4) * 8;
  v16h a;
#pragma unroll
  for (int v = 0; v < 8; ++v) {
    int k = k0 + ((v < 4) ? (kb + 2 * v) : (16 + kb + 2 * (v - 4)));
    a[2 * v]     = (_Float16)((okr && k     < Kr) ? src[k]     : 0.f);
    a[2 * v + 1] = (_Float16)((okr && k + 1 < Kr) ? src[k + 1] : 0.f);
  }
  return a;
}

// Guarded B fragment from f32 row-major weights/activations W [Wrows x ldw]
__device__ __forceinline__ v16h load_b_frag_wt(const float* W, int ldw, int n0, int k0,
                                               int Wrows, int Kr) {
  int lane = threadIdx.x & 31;
  int nn = n0 + (lane & 15);
  int kb = (lane >> 4) * 8;
  v16h b;
#pragma unroll
  for (int s = 0; s < 16; ++s) {
    int k = k0 + ((s < 8) ? (kb + s) : (16 + kb + (s - 8)));
    b[s] = (_Float16)((k < Kr && nn < Wrows) ? W[(long long)nn * ldw + k] : 0.f);
  }
  return b;
}

// A fragment from f16 LDS (row-major, ld = 32)
__device__ __forceinline__ v16h load_a_frag_lds(const _Float16* Q, int r0, int rmax) {
  int lane = threadIdx.x & 31;
  int row  = r0 + (lane & 15);
  bool okr = row < rmax;
  const _Float16* src = Q + row * 32;
  int kb = (lane >> 4) * 8;
  v16h a;
#pragma unroll
  for (int v = 0; v < 8; ++v) {
    int k = (v < 4) ? (kb + 2 * v) : (16 + kb + 2 * (v - 4));
    a[2 * v]     = okr ? src[k]     : (_Float16)0.f;
    a[2 * v + 1] = okr ? src[k + 1] : (_Float16)0.f;
  }
  return a;
}

// Combined A fragments for cross-attn: aV from mem, aK from mem+mem_pos.
__device__ __forceinline__ void load_mem_frags(const float* mem, const float* pos,
                                               int lda, long long r0, int k0,
                                               v16h* aV, v16h* aK) {
  int lane = threadIdx.x & 31;
  long long row = r0 + (lane & 15);
  const float* s1 = mem + row * lda + k0 + ((lane >> 4) * 8);
  const float* s2 = pos + row * lda + k0 + ((lane >> 4) * 8);
  v4f m0 = *(const v4f*)(s1);
  v4f m1 = *(const v4f*)(s1 + 4);
  v4f m2 = *(const v4f*)(s1 + 16);
  v4f m3 = *(const v4f*)(s1 + 20);
  v4f p0 = *(const v4f*)(s2);
  v4f p1 = *(const v4f*)(s2 + 4);
  v4f p2 = *(const v4f*)(s2 + 16);
  v4f p3 = *(const v4f*)(s2 + 20);
  v4f k0v = m0 + p0, k1v = m1 + p1, k2v = m2 + p2, k3v = m3 + p3;
  v16h av, ak;
#pragma unroll
  for (int i = 0; i < 4; ++i) {
    av[i]      = (_Float16)m0[i];  ak[i]      = (_Float16)k0v[i];
    av[4 + i]  = (_Float16)m1[i];  ak[4 + i]  = (_Float16)k1v[i];
    av[8 + i]  = (_Float16)m2[i];  ak[8 + i]  = (_Float16)k2v[i];
    av[12 + i] = (_Float16)m3[i];  ak[12 + i] = (_Float16)k3v[i];
  }
  *aV = av; *aK = ak;
}

// ---------------------------------------------------------------------------
// Fast GEMM: out = act(A @ W^T + bias), A f32, weights pre-transposed f16
// Wt [Kr x ldn]. One wave per 16x64 output (4 WMMA tiles share one A frag).
// ---------------------------------------------------------------------------
__device__ __forceinline__ void store_tile(float* __restrict__ out, int ldo, int r0,
                                           int col, const v8f& c, const float* bias,
                                           int relu) {
  int lane = threadIdx.x & 31;
  int hi8 = (lane >> 4) * 8;
  float bvv = bias ? bias[col] : 0.f;
#pragma unroll
  for (int r = 0; r < 8; ++r) {
    float v = c[r] + bvv;
    if (relu) v = fmaxf(v, 0.f);
    out[(long long)(r0 + r + hi8) * ldo + col] = v;
  }
}

__global__ __launch_bounds__(32) void wmma_gemm_fast(
    const float* __restrict__ A, int lda,
    const _Float16* __restrict__ Wt, int ldn,
    const float* __restrict__ bias,
    float* __restrict__ out, int ldo,
    int Kr, int relu) {
  int r0 = blockIdx.x * 16;
  int n0 = blockIdx.y * 64;
  v8f c0 = {}, c1 = {}, c2 = {}, c3 = {};
  for (int k0 = 0; k0 < Kr; k0 += 32) {
    if (k0 + 32 < Kr) {
      __builtin_prefetch(A + (long long)(r0 + (threadIdx.x & 15)) * lda + k0 + 32, 0, 1);
    }
    v16h a  = load_a_frag_fast(A, lda, r0, k0);
    v16h b0 = load_b_frag_t(Wt, ldn, n0,      k0);
    v16h b1 = load_b_frag_t(Wt, ldn, n0 + 16, k0);
    v16h b2 = load_b_frag_t(Wt, ldn, n0 + 32, k0);
    v16h b3 = load_b_frag_t(Wt, ldn, n0 + 48, k0);
    c0 = __builtin_amdgcn_wmma_f32_16x16x32_f16(false, a, false, b0, (short)0, c0, false, false);
    c1 = __builtin_amdgcn_wmma_f32_16x16x32_f16(false, a, false, b1, (short)0, c1, false, false);
    c2 = __builtin_amdgcn_wmma_f32_16x16x32_f16(false, a, false, b2, (short)0, c2, false, false);
    c3 = __builtin_amdgcn_wmma_f32_16x16x32_f16(false, a, false, b3, (short)0, c3, false, false);
  }
  int colm = threadIdx.x & 15;
  store_tile(out, ldo, r0, n0 + colm,      c0, bias, relu);
  store_tile(out, ldo, r0, n0 + 16 + colm, c1, bias, relu);
  store_tile(out, ldo, r0, n0 + 32 + colm, c2, bias, relu);
  store_tile(out, ldo, r0, n0 + 48 + colm, c3, bias, relu);
}

// Guarded batched GEMM for ragged cases (importance, K=8 spatial MLP)
__global__ __launch_bounds__(32) void wmma_gemm_nt(
    const float* __restrict__ A, int lda, long long aB,
    const float* __restrict__ W, int ldw, long long wB, int wRows,
    const float* __restrict__ bias,
    float* __restrict__ out, int ldo, long long oB,
    int Mr, int Nr, int Kr, int relu) {
  int bz = blockIdx.z;
  const float* Ab = A + (long long)bz * aB;
  const float* Wb = W + (long long)bz * wB;
  float*       Ob = out + (long long)bz * oB;
  int r0 = blockIdx.x * 16, n0 = blockIdx.y * 16;
  v8f c = {};
  for (int k0 = 0; k0 < Kr; k0 += 32) {
    v16h a = load_a_frag(Ab, lda, r0, k0, Mr, Kr);
    v16h b = load_b_frag_wt(Wb, ldw, n0, k0, wRows, Kr);
    c = __builtin_amdgcn_wmma_f32_16x16x32_f16(false, a, false, b, (short)0, c, false, false);
  }
  int lane = threadIdx.x & 31;
  int col  = n0 + (lane & 15);
  if (col >= Nr) return;
  float bvv = bias ? bias[col] : 0.f;
  int hi8 = (lane >> 4) * 8;
#pragma unroll
  for (int r = 0; r < 8; ++r) {
    int row = r0 + r + hi8;
    if (row < Mr) {
      float v = c[r] + bvv;
      if (relu) v = fmaxf(v, 0.f);
      Ob[(long long)row * ldo + col] = v;
    }
  }
}

// ---------------------------------------------------------------------------
// Weight transpose + f16 convert: Wt[l][k][n] = (f16) W[l][n][k]
// ---------------------------------------------------------------------------
__global__ void transpose_f16_kernel(const float* __restrict__ W, _Float16* __restrict__ Wt,
                                     int N, int K, int total) {
  int t = blockIdx.x * blockDim.x + threadIdx.x;
  if (t >= total) return;
  int nk = N * K;
  int l = t / nk, r = t - l * nk;
  int k = r / N, n = r - k * N;
  Wt[t] = (_Float16)W[(long long)l * nk + (long long)n * K + k];
}

// ---------------------------------------------------------------------------
// Row normalize: x /= (||x|| + 1e-6), row length 256
// ---------------------------------------------------------------------------
__global__ __launch_bounds__(256) void rownorm_kernel(float* __restrict__ x) {
  int row = blockIdx.x, tid = threadIdx.x;
  float v = x[(long long)row * 256 + tid];
  float ss = v * v;
#pragma unroll
  for (int m = 16; m > 0; m >>= 1) ss += __shfl_xor(ss, m, 32);
  __shared__ float red[8];
  if ((tid & 31) == 0) red[tid >> 5] = ss;
  __syncthreads();
  float tot = red[0] + red[1] + red[2] + red[3] + red[4] + red[5] + red[6] + red[7];
  x[(long long)row * 256 + tid] = v / (sqrtf(tot) + 1e-6f);
}

// ---------------------------------------------------------------------------
// Top-K (K=100) per batch, exact jax.lax.top_k order. The 40KB importance
// slab is pulled into LDS by the Tensor Data Mover (1 descriptor), then the
// iterative selection runs LDS-resident.
// ---------------------------------------------------------------------------
__global__ __launch_bounds__(256) void topk_kernel(const float* __restrict__ imp,
                                                   int* __restrict__ idx_out) {
  __shared__ float vals[10000];
  __shared__ float bv[256];
  __shared__ int   bi[256];
  int b = blockIdx.x, tid = threadIdx.x;
  const float* src = imp + (long long)b * 10000;
  if (tid < 32) {  // wave 0 issues the DMA and waits on TENSORcnt
    tdm_load_2d_f32((uint32_t)(uintptr_t)&vals[0], src,
                    /*tile*/ 10000, 1, /*tensor*/ 10000, 1, /*stride*/ 10000);
    __builtin_amdgcn_s_wait_tensorcnt(0);
  }
  __syncthreads();
  for (int k = 0; k < 100; ++k) {
    float best = -3.4e38f;
    int   besti = 0x7fffffff;
    for (int i = tid; i < 10000; i += 256) {
      float v = vals[i];
      if (v > best || (v == best && i < besti)) { best = v; besti = i; }
    }
    bv[tid] = best; bi[tid] = besti;
    __syncthreads();
    for (int s = 128; s > 0; s >>= 1) {
      if (tid < s) {
        float v2 = bv[tid + s]; int i2 = bi[tid + s];
        if (v2 > bv[tid] || (v2 == bv[tid] && i2 < bi[tid])) { bv[tid] = v2; bi[tid] = i2; }
      }
      __syncthreads();
    }
    if (tid == 0) { idx_out[b * 100 + k] = bi[0]; vals[bi[0]] = -3.4e38f; }
    __syncthreads();
  }
}

// ---------------------------------------------------------------------------
// Gathers / elementwise utilities
// ---------------------------------------------------------------------------
__global__ __launch_bounds__(256) void pair_gather(const float* __restrict__ hs,
                                                   const int* __restrict__ idx,
                                                   float* __restrict__ paircat) {
  int bk = blockIdx.x;
  int b = bk / 100;
  int id = idx[bk];
  int sp = id / 100, op = id % 100;
  const float* srow = hs + (long long)(b * 200 + 2 * sp)     * 256;
  const float* orow = hs + (long long)(b * 200 + 2 * op + 1) * 256;
  float* dst = paircat + (long long)bk * 512;
  for (int c = threadIdx.x; c < 256; c += blockDim.x) {
    dst[c]       = srow[c];
    dst[256 + c] = orow[c];
  }
}

__global__ void box_gather(const float* __restrict__ sb, const float* __restrict__ ob,
                           float* __restrict__ boxcat, int n) {
  int t = blockIdx.x * blockDim.x + threadIdx.x;
  if (t >= n) return;
  int row = t >> 3, j = t & 7;
  boxcat[t] = (j < 4) ? sb[row * 4 + j] : ob[row * 4 + j - 4];
}

__global__ void add_kernel(float* __restrict__ o, const float* __restrict__ a,
                           const float* __restrict__ b, int n) {
  int t = blockIdx.x * blockDim.x + threadIdx.x;
  if (t < n) o[t] = a[t] + b[t];
}

__global__ void copy_kernel(float* __restrict__ o, const float* __restrict__ a, int n) {
  int t = blockIdx.x * blockDim.x + threadIdx.x;
  if (t < n) o[t] = a[t];
}

// ---------------------------------------------------------------------------
// q = LayerNorm(q + resid) * g + b   (row length 256)
// ---------------------------------------------------------------------------
__global__ __launch_bounds__(256) void add_ln_kernel(float* __restrict__ q,
                                                     const float* __restrict__ resid,
                                                     const float* __restrict__ g,
                                                     const float* __restrict__ b_) {
  int row = blockIdx.x, tid = threadIdx.x;
  long long o = (long long)row * 256 + tid;
  float x = q[o] + resid[o];
  float s = x, ss = x * x;
#pragma unroll
  for (int m = 16; m > 0; m >>= 1) { s += __shfl_xor(s, m, 32); ss += __shfl_xor(ss, m, 32); }
  __shared__ float rs[8], rss[8];
  if ((tid & 31) == 0) { rs[tid >> 5] = s; rss[tid >> 5] = ss; }
  __syncthreads();
  float ts = 0.f, tss = 0.f;
#pragma unroll
  for (int i = 0; i < 8; ++i) { ts += rs[i]; tss += rss[i]; }
  float mean = ts * (1.f / 256.f);
  float var  = tss * (1.f / 256.f) - mean * mean;
  q[o] = (x - mean) * rsqrtf(var + 1e-5f) * g[tid] + b_[tid];
}

// ---------------------------------------------------------------------------
// Self-attention over 100 queries (dh=32). K/V head tiles (100x32, row
// stride 768 floats) are staged into LDS by two strided 2D TDM descriptors.
// ---------------------------------------------------------------------------
__global__ __launch_bounds__(128) void self_attn_kernel(const float* __restrict__ qkv,
                                                        float* __restrict__ outp) {
  int bh = blockIdx.x;
  int b = bh >> 3, h = bh & 7;
  __shared__ float kh[100][32];
  __shared__ float vh[100][32];
  int tid = threadIdx.x;
  const float* kbase = qkv + (long long)(b * 100) * 768 + 256 + h * 32;
  const float* vbase = qkv + (long long)(b * 100) * 768 + 512 + h * 32;
  if (tid < 32) {  // wave 0 issues both DMAs and waits on TENSORcnt
    tdm_load_2d_f32((uint32_t)(uintptr_t)&kh[0][0], kbase,
                    /*tile*/ 32, 100, /*tensor*/ 32, 100, /*stride*/ 768);
    tdm_load_2d_f32((uint32_t)(uintptr_t)&vh[0][0], vbase,
                    /*tile*/ 32, 100, /*tensor*/ 32, 100, /*stride*/ 768);
    __builtin_amdgcn_s_wait_tensorcnt(0);
  }
  __syncthreads();
  if (tid < 100) {
    const float* base = qkv + (long long)(b * 100 + tid) * 768;
    float qv[32];
#pragma unroll
    for (int d = 0; d < 32; ++d) qv[d] = base[h * 32 + d] * 0.17677669529663689f;
    float mx = -1e30f;
    for (int n = 0; n < 100; ++n) {
      float s = 0.f;
#pragma unroll
      for (int d = 0; d < 32; ++d) s += qv[d] * kh[n][d];
      mx = fmaxf(mx, s);
    }
    float sum = 0.f;
    float acc[32];
#pragma unroll
    for (int d = 0; d < 32; ++d) acc[d] = 0.f;
    for (int n = 0; n < 100; ++n) {
      float s = 0.f;
#pragma unroll
      for (int d = 0; d < 32; ++d) s += qv[d] * kh[n][d];
      float p = __expf(s - mx);
      sum += p;
#pragma unroll
      for (int d = 0; d < 32; ++d) acc[d] += p * vh[n][d];
    }
    float inv = 1.f / sum;
    float* orow = outp + (long long)(b * 100 + tid) * 256 + h * 32;
#pragma unroll
    for (int d = 0; d < 32; ++d) orow[d] = acc[d] * inv;
  }
}

// ---------------------------------------------------------------------------
// Cross-attention, flash-style with fused WMMA K/V projection.
// ---------------------------------------------------------------------------
__global__ __launch_bounds__(256) void cross_attn_kernel(
    const float* __restrict__ q,         // (3200 x 256)
    const float* __restrict__ mem,       // (B*4096 x 256)
    const float* __restrict__ mem_pos,   // (B*4096 x 256)
    const float* __restrict__ in_w,      // (768 x 256) f32 layer slice (q proj)
    const _Float16* __restrict__ in_wt,  // (256 x 768) f16 transposed layer slice
    const float* __restrict__ in_b,      // (768)
    float* __restrict__ outp) {          // (3200 x 256) concat-head
  int bh = blockIdx.x;
  int b = bh >> 3, h = bh & 7;
  int tid = threadIdx.x, lane = tid & 31, w = tid >> 5;

  __shared__ __align__(32) _Float16 q_lds[100 * 32];
  __shared__ __align__(32) _Float16 k_lds[8][512];  // [chunk][d*16 + m] (transposed)
  __shared__ float v_lds[8][512];                   // [chunk][m*32 + d]
  __shared__ float p_lds[7][16][16];
  __shared__ float sm[7][16], sl[7][16], ssc[7][16];

  // q head projection: q_h = (q @ wq_h^T + bq) / sqrt(dh), f16 into LDS
  const float* wq = in_w + (long long)(h * 32) * 256;
  const float* bq = in_b + h * 32;
  for (int t = tid; t < 100 * 32; t += 256) {
    int n = t >> 5, d = t & 31;
    const v4f* qr = (const v4f*)(q + (long long)(b * 100 + n) * 256);
    const v4f* wr = (const v4f*)(wq + (long long)d * 256);
    float s = 0.f;
#pragma unroll 8
    for (int c = 0; c < 64; ++c) {
      v4f a = qr[c], ww = wr[c];
      s += a[0] * ww[0] + a[1] * ww[1] + a[2] * ww[2] + a[3] * ww[3];
    }
    q_lds[n * 32 + d] = (_Float16)((s + bq[d]) * 0.17677669529663689f);
  }
  if (tid < 112) {
    int ww = tid >> 4, rr = tid & 15;
    sm[ww][rr] = -1e30f; sl[ww][rr] = 0.f; ssc[ww][rr] = 0.f;
  }
  __syncthreads();

  v16h aq = {};
  if (w < 7) aq = load_a_frag_lds(q_lds, w * 16, 100);
  float acc[16];
#pragma unroll
  for (int i = 0; i < 16; ++i) acc[i] = 0.f;

  int colm = lane & 15;
  int hi8  = (lane >> 4) * 8;
  const _Float16* wkt = in_wt + 256 + h * 32;  // + k*768 later
  const _Float16* wvt = in_wt + 512 + h * 32;
  float bk0 = in_b[256 + h * 32 + colm],      bk1 = in_b[256 + h * 32 + 16 + colm];
  float bv0 = in_b[512 + h * 32 + colm],      bv1 = in_b[512 + h * 32 + 16 + colm];

  for (int it = 0; it < 4096 / 128; ++it) {
    // ---- producer: each wave projects its own 16-row memory chunk ----
    long long mrow0 = (long long)b * 4096 + it * 128 + w * 16;
    v8f cK0 = {}, cK1 = {}, cV0 = {}, cV1 = {};
    for (int k0 = 0; k0 < 256; k0 += 32) {
      v16h aV, aK;
      load_mem_frags(mem, mem_pos, 256, mrow0, k0, &aV, &aK);
      v16h bK0f = load_b_frag_t(wkt, 768, 0, k0);
      v16h bK1f = load_b_frag_t(wkt, 768, 16, k0);
      v16h bV0f = load_b_frag_t(wvt, 768, 0, k0);
      v16h bV1f = load_b_frag_t(wvt, 768, 16, k0);
      cK0 = __builtin_amdgcn_wmma_f32_16x16x32_f16(false, aK, false, bK0f, (short)0, cK0, false, false);
      cK1 = __builtin_amdgcn_wmma_f32_16x16x32_f16(false, aK, false, bK1f, (short)0, cK1, false, false);
      cV0 = __builtin_amdgcn_wmma_f32_16x16x32_f16(false, aV, false, bV0f, (short)0, cV0, false, false);
      cV1 = __builtin_amdgcn_wmma_f32_16x16x32_f16(false, aV, false, bV1f, (short)0, cV1, false, false);
    }
#pragma unroll
    for (int r = 0; r < 8; ++r) {
      int m = r + hi8;
      k_lds[w][colm * 16 + m]        = (_Float16)(cK0[r] + bk0);
      k_lds[w][(16 + colm) * 16 + m] = (_Float16)(cK1[r] + bk1);
      v_lds[w][m * 32 + colm]        = cV0[r] + bv0;
      v_lds[w][m * 32 + 16 + colm]   = cV1[r] + bv1;
    }
    __syncthreads();

    // ---- consumer: waves 0..6 own 16 query rows each ----
    if (w < 7) {
      for (int j = 0; j < 8; ++j) {
        v16h bf;
#pragma unroll
        for (int s2 = 0; s2 < 16; ++s2)
          bf[s2] = k_lds[j][((s2 < 8) ? (hi8 + s2) : (16 + hi8 + (s2 - 8))) * 16 + colm];
        v8f cz = {};
        v8f s = __builtin_amdgcn_wmma_f32_16x16x32_f16(false, aq, false, bf, (short)0, cz, false, false);
        float tmax[8], p[8];
#pragma unroll
        for (int r = 0; r < 8; ++r) {
          float t = s[r];
          t = fmaxf(t, __shfl_xor(t, 1, 32));
          t = fmaxf(t, __shfl_xor(t, 2, 32));
          t = fmaxf(t, __shfl_xor(t, 4, 32));
          t = fmaxf(t, __shfl_xor(t, 8, 32));
          tmax[r] = t;
        }
#pragma unroll
        for (int r = 0; r < 8; ++r) {
          int row = r + hi8;
          float m_old = sm[w][row];
          float m_new = fmaxf(m_old, tmax[r]);
          p[r] = __expf(s[r] - m_new);
          float z = p[r];
          z += __shfl_xor(z, 1, 32);
          z += __shfl_xor(z, 2, 32);
          z += __shfl_xor(z, 4, 32);
          z += __shfl_xor(z, 8, 32);
          if (colm == 0) {
            float sc = __expf(m_old - m_new);
            ssc[w][row] = sc;
            sl[w][row]  = sl[w][row] * sc + z;
            sm[w][row]  = m_new;
          }
          p_lds[w][row][colm] = p[r];
        }
        int ql = colm, dh2 = lane >> 4;
        float sc = ssc[w][ql];
#pragma unroll
        for (int dd = 0; dd < 16; ++dd) acc[dd] *= sc;
        for (int m2 = 0; m2 < 16; ++m2) {
          float pp = p_lds[w][ql][m2];
          const float* vr = &v_lds[j][m2 * 32 + dh2 * 16];
#pragma unroll
          for (int dd = 0; dd < 16; ++dd) acc[dd] += pp * vr[dd];
        }
      }
    }
    __syncthreads();
  }

  if (w < 7) {
    int ql = colm, dh2 = lane >> 4;
    int qg = w * 16 + ql;
    if (qg < 100) {
      float inv = 1.f / sl[w][ql];
      float* orow = outp + (long long)(b * 100 + qg) * 256 + h * 32 + dh2 * 16;
#pragma unroll
      for (int dd = 0; dd < 16; ++dd) orow[dd] = acc[dd] * inv;
    }
  }
}

// ---------------------------------------------------------------------------
// Host orchestration
// ---------------------------------------------------------------------------
extern "C" void kernel_launch(void* const* d_in, const int* in_sizes, int n_in,
                              void* d_out, int out_size, void* d_ws, size_t ws_size,
                              hipStream_t stream) {
  (void)in_sizes; (void)n_in; (void)out_size; (void)ws_size;
  const float* hs       = (const float*)d_in[0];
  const float* mem      = (const float*)d_in[1];
  const float* mem_pos  = (const float*)d_in[2];
  const float* sub_box  = (const float*)d_in[3];
  const float* obj_box  = (const float*)d_in[4];
  const float* su_w1 = (const float*)d_in[5];  const float* su_b1 = (const float*)d_in[6];
  const float* su_w2 = (const float*)d_in[7];  const float* su_b2 = (const float*)d_in[8];
  const float* ou_w1 = (const float*)d_in[9];  const float* ou_b1 = (const float*)d_in[10];
  const float* ou_w2 = (const float*)d_in[11]; const float* ou_b2 = (const float*)d_in[12];
  const float* pm_w1 = (const float*)d_in[13]; const float* pm_b1 = (const float*)d_in[14];
  const float* pm_w2 = (const float*)d_in[15]; const float* pm_b2 = (const float*)d_in[16];
  const float* sp_w1 = (const float*)d_in[17]; const float* sp_b1 = (const float*)d_in[18];
  const float* sp_w2 = (const float*)d_in[19]; const float* sp_b2 = (const float*)d_in[20];
  const float* sa_in_w  = (const float*)d_in[21]; const float* sa_in_b  = (const float*)d_in[22];
  const float* sa_out_w = (const float*)d_in[23]; const float* sa_out_b = (const float*)d_in[24];
  const float* ca_in_w  = (const float*)d_in[25]; const float* ca_in_b  = (const float*)d_in[26];
  const float* ca_out_w = (const float*)d_in[27]; const float* ca_out_b = (const float*)d_in[28];
  const float* l1_w = (const float*)d_in[29]; const float* l1_b = (const float*)d_in[30];
  const float* l2_w = (const float*)d_in[31]; const float* l2_b = (const float*)d_in[32];
  const float* n1_g = (const float*)d_in[33]; const float* n1_b = (const float*)d_in[34];
  const float* n2_g = (const float*)d_in[35]; const float* n2_b = (const float*)d_in[36];
  const float* n3_g = (const float*)d_in[37]; const float* n3_b = (const float*)d_in[38];

  // ---- workspace layout: f32 activations (~67 MB) + f16 transposed weights (~20 MB)
  float* ws = (float*)d_ws;
  size_t o = 0;
  float* sub_emb = ws + o; o += 819200;
  float* obj_emb = ws + o; o += 819200;
  float* imp     = ws + o; o += 320000;
  int*   idxbuf  = (int*)(ws + o); o += 3200;
  float* paircat = ws + o; o += 1638400;
  float* spbuf   = ws + o; o += 819200;
  float* qbuf    = ws + o; o += 819200;
  float* qsp     = ws + o; o += 819200;
  float* qkv     = ws + o; o += 2457600;
  float* attnout = ws + o; o += 819200;
  float* projbuf = ws + o; o += 819200;
  float* hidden  = ws + o; o += 6553600;
  float* boxcat  = ws + o; o += 25600;

  _Float16* h16 = (_Float16*)(ws + o);
  size_t ho = 0;
  _Float16* su_w1t = h16 + ho; ho += 65536;
  _Float16* su_w2t = h16 + ho; ho += 65536;
  _Float16* ou_w1t = h16 + ho; ho += 65536;
  _Float16* ou_w2t = h16 + ho; ho += 65536;
  _Float16* pm_w1t = h16 + ho; ho += 131072;
  _Float16* pm_w2t = h16 + ho; ho += 65536;
  _Float16* sp_w2t = h16 + ho; ho += 65536;
  _Float16* sa_in_t  = h16 + ho; ho += 1179648;   // 6 x (256 x 768)
  _Float16* sa_out_t = h16 + ho; ho += 393216;    // 6 x (256 x 256)
  _Float16* ca_in_t  = h16 + ho; ho += 1179648;
  _Float16* ca_out_t = h16 + ho; ho += 393216;
  _Float16* l1_t     = h16 + ho; ho += 3145728;   // 6 x (256 x 2048)
  _Float16* l2_t     = h16 + ho; ho += 3145728;   // 6 x (2048 x 256)

  auto tr = [&](const float* W, _Float16* Wt, int N, int K, int batch) {
    int total = batch * N * K;
    transpose_f16_kernel<<<(total + 255) / 256, 256, 0, stream>>>(W, Wt, N, K, total);
  };
  tr(su_w1, su_w1t, 256, 256, 1);   tr(su_w2, su_w2t, 256, 256, 1);
  tr(ou_w1, ou_w1t, 256, 256, 1);   tr(ou_w2, ou_w2t, 256, 256, 1);
  tr(pm_w1, pm_w1t, 256, 512, 1);   tr(pm_w2, pm_w2t, 256, 256, 1);
  tr(sp_w2, sp_w2t, 256, 256, 1);
  tr(sa_in_w,  sa_in_t,  768, 256, 6);   tr(sa_out_w, sa_out_t, 256, 256, 6);
  tr(ca_in_w,  ca_in_t,  768, 256, 6);   tr(ca_out_w, ca_out_t, 256, 256, 6);
  tr(l1_w, l1_t, 2048, 256, 6);          tr(l2_w, l2_t, 256, 2048, 6);

  // fast GEMM: M fixed 3200, N multiple of 64
  auto gemmf = [&](const float* A, int lda, const _Float16* Wt, int ldn,
                   const float* bias, float* out, int Nr, int Kr, int relu) {
    dim3 g(200, (unsigned)(Nr / 64), 1);
    wmma_gemm_fast<<<g, 32, 0, stream>>>(A, lda, Wt, ldn, bias, out, Nr, Kr, relu);
  };

  // ---- pair embeddings + normalize ----
  gemmf(hs,       512, su_w1t, 256, su_b1, hidden,  256, 256, 1);
  gemmf(hidden,   256, su_w2t, 256, su_b2, sub_emb, 256, 256, 0);
  gemmf(hs + 256, 512, ou_w1t, 256, ou_b1, hidden,  256, 256, 1);
  gemmf(hidden,   256, ou_w2t, 256, ou_b2, obj_emb, 256, 256, 0);
  rownorm_kernel<<<3200, 256, 0, stream>>>(sub_emb);
  rownorm_kernel<<<3200, 256, 0, stream>>>(obj_emb);

  // ---- importance (batched sub_emb @ obj_emb^T, ragged 100x100) + top-K ----
  wmma_gemm_nt<<<dim3(7, 7, 32), 32, 0, stream>>>(
      sub_emb, 256, 25600, obj_emb, 256, 25600, 100, nullptr,
      imp, 100, 10000, 100, 100, 256, 0);
  topk_kernel<<<32, 256, 0, stream>>>(imp, idxbuf);

  // ---- gather + pair / spatial MLPs ----
  pair_gather<<<3200, 256, 0, stream>>>(hs, idxbuf, paircat);
  box_gather<<<(25600 + 255) / 256, 256, 0, stream>>>(sub_box, obj_box, boxcat, 25600);
  gemmf(paircat, 512, pm_w1t, 256, pm_b1, hidden, 256, 512, 1);
  gemmf(hidden,  256, pm_w2t, 256, pm_b2, qbuf,   256, 256, 0);
  wmma_gemm_nt<<<dim3(200, 16, 1), 32, 0, stream>>>(   // K=8 ragged
      boxcat, 8, 0, sp_w1, 8, 0, 256, sp_b1, hidden, 256, 0, 3200, 256, 8, 1);
  gemmf(hidden, 256, sp_w2t, 256, sp_b2, spbuf, 256, 256, 0);

  // ---- 6 decoder layers ----
  for (int i = 0; i < 6; ++i) {
    const float*    ca_in_w_i = ca_in_w + (long long)i * 768 * 256;
    const float*    ca_in_b_i = ca_in_b + (long long)i * 768;
    const _Float16* ca_in_t_i = ca_in_t + (long long)i * 196608;

    // self-attention block
    add_kernel<<<(819200 + 255) / 256, 256, 0, stream>>>(qsp, qbuf, spbuf, 819200);
    gemmf(qsp, 256, sa_in_t + (long long)i * 196608, 768, sa_in_b + i * 768, qkv, 768, 256, 0);
    self_attn_kernel<<<256, 128, 0, stream>>>(qkv, attnout);
    gemmf(attnout, 256, sa_out_t + (long long)i * 65536, 256, sa_out_b + i * 256,
          projbuf, 256, 256, 0);
    add_ln_kernel<<<3200, 256, 0, stream>>>(qbuf, projbuf, n1_g + i * 256, n1_b + i * 256);

    // cross-attention block (fused KV projection, flash softmax)
    cross_attn_kernel<<<256, 256, 0, stream>>>(qbuf, mem, mem_pos, ca_in_w_i, ca_in_t_i,
                                               ca_in_b_i, attnout);
    gemmf(attnout, 256, ca_out_t + (long long)i * 65536, 256, ca_out_b + i * 256,
          projbuf, 256, 256, 0);
    add_ln_kernel<<<3200, 256, 0, stream>>>(qbuf, projbuf, n2_g + i * 256, n2_b + i * 256);

    // FFN block
    gemmf(qbuf, 256, l1_t + (long long)i * 524288, 2048, l1_b + i * 2048, hidden, 2048, 256, 1);
    gemmf(hidden, 2048, l2_t + (long long)i * 524288, 256, l2_b + i * 256, projbuf, 256, 2048, 0);
    add_ln_kernel<<<3200, 256, 0, stream>>>(qbuf, projbuf, n3_g + i * 256, n3_b + i * 256);
  }

  copy_kernel<<<(819200 + 255) / 256, 256, 0, stream>>>((float*)d_out, qbuf, 819200);
}
